// Attention_1520418422787
// MI455X (gfx1250) — matrix operand order repaired and checked by reading, not compiled, not run on hardware
//
#include <hip/hip_runtime.h>
#include <hip/hip_bf16.h>
#include <cstdint>

// ---------------------------------------------------------------------------
// Attention block for MI455X (gfx1250, wave32, WMMA + async global->LDS).
// B=4, S=2048, DIM=2048, H=16, KVH=4, HD=128.
// Pipeline: f32->f16 converts -> QKV GEMM (WMMA, async double-buffered) ->
//           RoPE+repack -> flash attention (WMMA QK^T + PV, async db) ->
//           output GEMM (WMMA).
// ---------------------------------------------------------------------------

typedef __attribute__((ext_vector_type(4)))  _Float16 v4h;
typedef __attribute__((ext_vector_type(8)))  _Float16 v8h;
typedef __attribute__((ext_vector_type(16))) _Float16 v16h;
typedef __attribute__((ext_vector_type(8)))  float    v8f;

#define BQ  4
#define SQ  2048
#define DIM 2048
#define NH  16
#define KVH 4
#define HD  128
#define NTOK (BQ * SQ)            // 8192
#define QKVN (DIM + 2 * KVH * HD) // 3072

// ---------------------------------------------------------------------------
// CDNA5 async copy: per-lane 16B global -> LDS, tracked by ASYNCcnt.
// GVS addressing: mem = SADDR(u64) + VADDR(i32). VDST = LDS byte address.
// ---------------------------------------------------------------------------
__device__ __forceinline__ void async_ld_b128(uint32_t lds_addr, uint32_t voff,
                                              const void* saddr) {
  asm volatile("global_load_async_to_lds_b128 %0, %1, %2"
               :
               : "v"(lds_addr), "v"(voff), "s"(saddr)
               : "memory");
}
#define WAIT_ASYNC(n) asm volatile("s_wait_asynccnt %0" ::"i"(n) : "memory")

// ---------------------------------------------------------------------------
// WMMA helpers. Register layouts per CDNA5 ISA 7.12.2 (16-bit, wave32):
//  A (16xK, k-step 32): lane<16 -> row=lane, halves = K {k0+hi*8..+7, k0+16+hi*8..+7}
//  B (Kx16):            lane<16 -> col=lane, halves = K {k0+hi*8..+7, k0+16+hi*8..+7}
//  C/D (16x16 f32):     lane<16 -> N=lane, VGPR i -> M=i ; lane>=16 -> M=8+i
// ---------------------------------------------------------------------------
__device__ __forceinline__ v8f wmma_f16(v16h a, v16h b, v8f c) {
  return __builtin_amdgcn_wmma_f32_16x16x32_f16(false, a, false, b, (short)0, c,
                                                false, false);
}

__device__ __forceinline__ v16h ld_frag_a(const _Float16* base, int stride, int lane) {
  const int r = lane & 15, hi = lane >> 4;
  const _Float16* p = base + r * stride + hi * 8;
  v16h o;
  ((v8h*)&o)[0] = *(const v8h*)(p);
  ((v8h*)&o)[1] = *(const v8h*)(p + 16);
  return o;
}

__device__ __forceinline__ v16h ld_frag_b(const _Float16* base, int stride, int lane) {
  const int r = lane & 15, hi = lane >> 4;
  const _Float16* p = base + r * stride + hi * 8;
  v16h o;
  ((v8h*)&o)[0] = *(const v8h*)(p);
  ((v8h*)&o)[1] = *(const v8h*)(p + 16);
  return o;
}

// ---------------------------------------------------------------------------
// fp32 -> f16 conversion, 4 elements/thread
// ---------------------------------------------------------------------------
__global__ __launch_bounds__(256) void f32_to_f16(const float* __restrict__ s,
                                                  _Float16* __restrict__ d, int n) {
  const int i = (blockIdx.x * 256 + threadIdx.x) * 4;
  if (i + 3 < n) {
    const float4 v = *(const float4*)(s + i);
    v4h o;
    o.x = (_Float16)v.x; o.y = (_Float16)v.y;
    o.z = (_Float16)v.z; o.w = (_Float16)v.w;
    *(v4h*)(d + i) = o;
  }
}

// ---------------------------------------------------------------------------
// GEMM: C[M][N] = A[M][K] * Bt[N][K]^T   (A,Bt f16 row-major, C = OutT)
// Block tile 128x128, BK=64, 256 threads = 8 waves (2 M x 4 N groups),
// wave tile 64x32 -> 4x2 accumulators. Double-buffered async LDS tiles.
// ---------------------------------------------------------------------------
#define G_LDA 72  // 64 + 8 halves pad
template <typename OutT>
__global__ __launch_bounds__(256) void gemm_nt(const _Float16* __restrict__ A,
                                               const _Float16* __restrict__ Bt,
                                               OutT* __restrict__ C,
                                               int M, int N, int K) {
  __shared__ __align__(16) _Float16 As[2][128 * G_LDA];
  __shared__ __align__(16) _Float16 Bs[2][128 * G_LDA];

  const int tid = threadIdx.x;
  const int lane = tid & 31;
  const int wid = tid >> 5;
  const int wm = wid & 1;
  const int wn = wid >> 1;
  const int m0 = blockIdx.y * 128;
  const int n0 = blockIdx.x * 128;

  // per-thread async-load geometry (4 b128 segs per tile per thread)
  uint32_t ldsA0[4], ldsB0[4], voA[4], voB[4];
#pragma unroll
  for (int s = 0; s < 4; ++s) {
    const int sid = s * 256 + tid;
    const int r_ = sid >> 3, c_ = (sid & 7) << 3;
    ldsA0[s] = (uint32_t)(size_t)&As[0][r_ * G_LDA + c_];
    ldsB0[s] = (uint32_t)(size_t)&Bs[0][r_ * G_LDA + c_];
    voA[s] = (uint32_t)(((size_t)(m0 + r_) * K + c_) * 2);
    voB[s] = (uint32_t)(((size_t)(n0 + r_) * K + c_) * 2);
  }
  const uint32_t TILE_B = 128 * G_LDA * 2;  // bytes between buffers

  v8f acc[4][2];
#pragma unroll
  for (int i = 0; i < 4; ++i)
#pragma unroll
    for (int j = 0; j < 2; ++j)
#pragma unroll
      for (int e = 0; e < 8; ++e) acc[i][j][e] = 0.0f;

  // prologue: prefetch first tile into buffer 0
#pragma unroll
  for (int s = 0; s < 4; ++s) {
    async_ld_b128(ldsA0[s], voA[s], A);
    async_ld_b128(ldsB0[s], voB[s], Bt);
  }

  const int niter = K >> 6;
  for (int it = 0; it < niter; ++it) {
    const int buf = it & 1;
    if (it + 1 < niter) {
      const _Float16* Ak = A + (size_t)(it + 1) * 64;
      const _Float16* Bk = Bt + (size_t)(it + 1) * 64;
      const uint32_t boff = buf ? 0u : TILE_B;  // buf^1 base offset
#pragma unroll
      for (int s = 0; s < 4; ++s) {
        async_ld_b128(ldsA0[s] + boff, voA[s], Ak);
        async_ld_b128(ldsB0[s] + boff, voB[s], Bk);
      }
      WAIT_ASYNC(8);  // drain previous batch (ASYNCcnt completes in order)
    } else {
      WAIT_ASYNC(0);
    }
    __syncthreads();

    const _Float16* Ab = &As[buf][0];
    const _Float16* Bb = &Bs[buf][0];
#pragma unroll
    for (int kt = 0; kt < 2; ++kt) {
      const int k0 = kt * 32;
      v16h af[4], bf[2];
#pragma unroll
      for (int mi = 0; mi < 4; ++mi)
        af[mi] = ld_frag_a(Ab + (wm * 64 + mi * 16) * G_LDA + k0, G_LDA, lane);
#pragma unroll
      for (int ni = 0; ni < 2; ++ni)
        bf[ni] = ld_frag_b(Bb + (wn * 32 + ni * 16) * G_LDA + k0, G_LDA, lane);
#pragma unroll
      for (int mi = 0; mi < 4; ++mi)
#pragma unroll
        for (int ni = 0; ni < 2; ++ni)
          acc[mi][ni] = wmma_f16(af[mi], bf[ni], acc[mi][ni]);
    }
    __syncthreads();  // done reading buf before it is overwritten next round
  }

  const int r = lane & 15, hi = lane >> 4;
#pragma unroll
  for (int mi = 0; mi < 4; ++mi) {
#pragma unroll
    for (int ni = 0; ni < 2; ++ni) {
      const int cm = m0 + wm * 64 + mi * 16 + hi * 8;
      const int cn = n0 + wn * 32 + ni * 16 + r;
#pragma unroll
      for (int i = 0; i < 8; ++i)
        C[(size_t)(cm + i) * N + cn] = (OutT)acc[mi][ni][i];
    }
  }
}

// ---------------------------------------------------------------------------
// RoPE + repack. One block per token.
//   qkv16[tok][3072]: cols 0..2047 Q, 2048..2559 K, 2560..3071 V (f16)
// Outputs: q16 [B][16][S][128] (scaled 1/sqrt(128)), k16 [B][4][S][128],
//          vt16 [B][4][128][S] (transposed for PV B-fragments).
// ---------------------------------------------------------------------------
__global__ __launch_bounds__(256) void rope_repack(const _Float16* __restrict__ qkv,
                                                   const float* __restrict__ freqs,
                                                   _Float16* __restrict__ q16,
                                                   _Float16* __restrict__ k16,
                                                   _Float16* __restrict__ vt16) {
  const int tok = blockIdx.x;
  const int b = tok >> 11, s = tok & 2047;
  const int tid = threadIdx.x;
  __shared__ float cs[64], sn[64];
  if (tid < 64) {
    float f = freqs[(size_t)tok * 64 + tid];
    float si, co;
    sincosf(f, &si, &co);
    cs[tid] = co;
    sn[tid] = si;
  }
  __syncthreads();

  const _Float16* row = qkv + (size_t)tok * QKVN;
  const float scale = 0.08838834764831845f;  // 1/sqrt(128), folded into Q

#pragma unroll
  for (int p = tid; p < 1024; p += 256) {  // Q: 16 heads x 64 pairs
    const int hq = p >> 6, d = p & 63;
    const float x0 = (float)row[hq * 128 + 2 * d];
    const float x1 = (float)row[hq * 128 + 2 * d + 1];
    const float c = cs[d], si = sn[d];
    const size_t dst = (((size_t)(b * NH + hq) * SQ) + s) * HD + 2 * d;
    q16[dst]     = (_Float16)((x0 * c - x1 * si) * scale);
    q16[dst + 1] = (_Float16)((x0 * si + x1 * c) * scale);
  }
  {  // K: 4 heads x 64 pairs = 256 (one per thread)
    const int hk = tid >> 6, d = tid & 63;
    const float x0 = (float)row[DIM + hk * 128 + 2 * d];
    const float x1 = (float)row[DIM + hk * 128 + 2 * d + 1];
    const float c = cs[d], si = sn[d];
    const size_t dst = (((size_t)(b * KVH + hk) * SQ) + s) * HD + 2 * d;
    k16[dst]     = (_Float16)(x0 * c - x1 * si);
    k16[dst + 1] = (_Float16)(x0 * si + x1 * c);
  }
#pragma unroll
  for (int e = tid; e < 512; e += 256) {  // V transpose copy
    const int hv = e >> 7, d = e & 127;
    vt16[(((size_t)(b * KVH + hv) * HD) + d) * SQ + s] = row[DIM + 512 + e];
  }
}

// ---------------------------------------------------------------------------
// Flash attention. Grid (S/64, H, B), 128 threads = 4 waves x 16 query rows.
// Per j-step (Bc=64): scores = Q(16x128) x K^T -> online softmax -> P f16 via
// per-wave LDS (C-layout -> A-layout) -> out += P(16x64) x V(64x128).
// K / V^T tiles double-buffered with async global->LDS loads.
// ---------------------------------------------------------------------------
#define K_LDS 136  // 128 + 8 pad (halves)
#define V_LDS 72   // 64 + 8 pad
#define P_LDS 72   // 64 + 8 pad
__global__ __launch_bounds__(128) void attn_kernel(const _Float16* __restrict__ q16,
                                                   const _Float16* __restrict__ k16,
                                                   const _Float16* __restrict__ vt16,
                                                   _Float16* __restrict__ attn) {
  const int b = blockIdx.z, h = blockIdx.y, qblk = blockIdx.x;
  const int kvh = h >> 2;  // N_REP = 4
  const int tid = threadIdx.x, w = tid >> 5, lane = tid & 31;
  const int r = lane & 15, hi = lane >> 4;

  __shared__ __align__(16) _Float16 Ks[2][64 * K_LDS];   // [key][d]
  __shared__ __align__(16) _Float16 Vs[2][128 * V_LDS];  // [d][key]
  __shared__ __align__(16) _Float16 Ps[4 * 16 * P_LDS];  // per-wave P staging

  const _Float16* Qh = q16 + ((size_t)(b * NH + h) * SQ + qblk * 64 + w * 16) * HD;
  const _Float16* Kh = k16 + ((size_t)(b * KVH + kvh) * SQ) * HD;
  const _Float16* Vh = vt16 + ((size_t)(b * KVH + kvh) * HD) * SQ;
  _Float16* Pw = Ps + w * (16 * P_LDS);

  // per-thread async geometry (8 b128 segs per tile per thread)
  uint32_t ldsK0[8], ldsV0[8], voK[8], voV[8];
#pragma unroll
  for (int s = 0; s < 8; ++s) {
    const int sid = s * 128 + tid;
    const int kr = sid >> 4, kc = (sid & 15) << 3;
    ldsK0[s] = (uint32_t)(size_t)&Ks[0][kr * K_LDS + kc];
    voK[s] = (uint32_t)(((size_t)kr * HD + kc) * 2);
    const int vr = sid >> 3, vc = (sid & 7) << 3;
    ldsV0[s] = (uint32_t)(size_t)&Vs[0][vr * V_LDS + vc];
    voV[s] = (uint32_t)(((size_t)vr * SQ + vc) * 2);
  }
  const uint32_t KTILE_B = 64 * K_LDS * 2;
  const uint32_t VTILE_B = 128 * V_LDS * 2;

  // Preload Q fragments (16 rows x 128 = 4 k-steps), direct from global.
  v16h qa[4];
#pragma unroll
  for (int kt = 0; kt < 4; ++kt) qa[kt] = ld_frag_a(Qh + kt * 32, HD, lane);

  float m_i[8], l_i[8];
  v8f oacc[8];
#pragma unroll
  for (int i = 0; i < 8; ++i) {
    m_i[i] = -1e30f;
    l_i[i] = 0.0f;
#pragma unroll
    for (int e = 0; e < 8; ++e) oacc[i][e] = 0.0f;
  }

  // prologue: prefetch first K/V tiles into buffer 0
#pragma unroll
  for (int s = 0; s < 8; ++s) {
    async_ld_b128(ldsK0[s], voK[s], Kh);
    async_ld_b128(ldsV0[s], voV[s], Vh);
  }

  for (int j0 = 0; j0 < SQ; j0 += 64) {
    const int buf = (j0 >> 6) & 1;
    if (j0 + 64 < SQ) {
      const _Float16* Kn = Kh + (size_t)(j0 + 64) * HD;
      const _Float16* Vn = Vh + (size_t)(j0 + 64);
      const uint32_t kb = buf ? 0u : KTILE_B;
      const uint32_t vb = buf ? 0u : VTILE_B;
#pragma unroll
      for (int s = 0; s < 8; ++s) {
        async_ld_b128(ldsK0[s] + kb, voK[s], Kn);
        async_ld_b128(ldsV0[s] + vb, voV[s], Vn);
      }
      WAIT_ASYNC(16);  // drain previous tile pair
    } else {
      WAIT_ASYNC(0);
    }
    __syncthreads();

    const _Float16* Kb = &Ks[buf][0];
    const _Float16* Vb = &Vs[buf][0];

    // scores = Q x K^T : 4 n-tiles x 4 k-steps
    v8f sc[4];
#pragma unroll
    for (int nt = 0; nt < 4; ++nt) {
#pragma unroll
      for (int e = 0; e < 8; ++e) sc[nt][e] = 0.0f;
#pragma unroll
      for (int kt = 0; kt < 4; ++kt)
        sc[nt] = wmma_f16(qa[kt],
                          ld_frag_b(Kb + (nt * 16) * K_LDS + kt * 32, K_LDS, lane),
                          sc[nt]);
    }

    // online softmax (rows live across lanes 0-15 / 16-31; masks 1,2,4,8)
#pragma unroll
    for (int i = 0; i < 8; ++i) {
      float rm = fmaxf(fmaxf(sc[0][i], sc[1][i]), fmaxf(sc[2][i], sc[3][i]));
      rm = fmaxf(rm, __shfl_xor(rm, 1));
      rm = fmaxf(rm, __shfl_xor(rm, 2));
      rm = fmaxf(rm, __shfl_xor(rm, 4));
      rm = fmaxf(rm, __shfl_xor(rm, 8));
      const float mn = fmaxf(m_i[i], rm);
      const float corr = __expf(m_i[i] - mn);
      m_i[i] = mn;
#pragma unroll
      for (int t = 0; t < 8; ++t) oacc[t][i] *= corr;
      float rs = 0.0f;
      const int prow = hi * 8 + i;
#pragma unroll
      for (int nt = 0; nt < 4; ++nt) {
        const float p = __expf(sc[nt][i] - mn);
        rs += p;
        Pw[prow * P_LDS + nt * 16 + r] = (_Float16)p;
      }
      rs += __shfl_xor(rs, 1);
      rs += __shfl_xor(rs, 2);
      rs += __shfl_xor(rs, 4);
      rs += __shfl_xor(rs, 8);
      l_i[i] = l_i[i] * corr + rs;
    }
    __syncthreads();  // P C-layout stores -> A-layout loads (cross-lane)

    // out += P(16x64) x V(64x128) : 2 k-steps x 8 n-tiles
#pragma unroll
    for (int kt = 0; kt < 2; ++kt) {
      const v16h pa = ld_frag_a(Pw + kt * 32, P_LDS, lane);
#pragma unroll
      for (int nt = 0; nt < 8; ++nt)
        oacc[nt] = wmma_f16(pa,
                            ld_frag_b(Vb + (nt * 16) * V_LDS + kt * 32, V_LDS, lane),
                            oacc[nt]);
    }
    __syncthreads();  // done reading buf before next iteration overwrites peer
  }

  // epilogue: divide by l, write token-major f16 [B][S][H*HD]
  float inv[8];
#pragma unroll
  for (int i = 0; i < 8; ++i) inv[i] = 1.0f / l_i[i];
  const size_t obase = ((size_t)b * SQ) * DIM + (size_t)h * HD;
#pragma unroll
  for (int nt = 0; nt < 8; ++nt) {
    const int d = nt * 16 + r;
#pragma unroll
    for (int i = 0; i < 8; ++i) {
      const int srow = qblk * 64 + w * 16 + hi * 8 + i;
      attn[obase + (size_t)srow * DIM + d] = (_Float16)(oacc[nt][i] * inv[i]);
    }
  }
}

// ---------------------------------------------------------------------------
// Host launch. Inputs: x, freqs, wq, wk, wv, wo (all fp32). Output fp32.
// ---------------------------------------------------------------------------
extern "C" void kernel_launch(void* const* d_in, const int* in_sizes, int n_in,
                              void* d_out, int out_size, void* d_ws, size_t ws_size,
                              hipStream_t stream) {
  (void)in_sizes; (void)n_in; (void)out_size; (void)ws_size;
  const float* x  = (const float*)d_in[0];
  const float* fr = (const float*)d_in[1];
  const float* wq = (const float*)d_in[2];
  const float* wk = (const float*)d_in[3];
  const float* wv = (const float*)d_in[4];
  const float* wo = (const float*)d_in[5];

  // workspace layout (f16 elements)
  _Float16* x16    = (_Float16*)d_ws;             // 8192*2048
  _Float16* wqkv16 = x16 + (size_t)NTOK * DIM;    // 3072*2048
  _Float16* wo16   = wqkv16 + (size_t)QKVN * DIM; // 2048*2048
  _Float16* qkv16  = wo16 + (size_t)DIM * DIM;    // 8192*3072
  _Float16* q16    = qkv16 + (size_t)NTOK * QKVN; // 8192*2048
  _Float16* k16    = q16 + (size_t)NTOK * DIM;    // 8192*512
  _Float16* vt16   = k16 + (size_t)NTOK * KVH * HD;
  _Float16* attn16 = vt16 + (size_t)NTOK * KVH * HD;

  f32_to_f16<<<(NTOK * DIM) / 1024, 256, 0, stream>>>(x, x16, NTOK * DIM);
  f32_to_f16<<<(DIM * DIM) / 1024, 256, 0, stream>>>(wq, wqkv16, DIM * DIM);
  f32_to_f16<<<(512 * DIM) / 1024, 256, 0, stream>>>(wk, wqkv16 + (size_t)DIM * DIM, 512 * DIM);
  f32_to_f16<<<(512 * DIM) / 1024, 256, 0, stream>>>(wv, wqkv16 + (size_t)(DIM + 512) * DIM, 512 * DIM);
  f32_to_f16<<<(DIM * DIM) / 1024, 256, 0, stream>>>(wo, wo16, DIM * DIM);

  {  // fused QKV projection: [8192,2048] x [3072,2048]^T -> f16
    dim3 g(QKVN / 128, NTOK / 128);
    gemm_nt<_Float16><<<g, 256, 0, stream>>>(x16, wqkv16, qkv16, NTOK, QKVN, DIM);
  }

  rope_repack<<<NTOK, 256, 0, stream>>>(qkv16, fr, q16, k16, vt16);

  {
    dim3 g(SQ / 64, NH, BQ);
    attn_kernel<<<g, 128, 0, stream>>>(q16, k16, vt16, attn16);
  }

  {  // output projection: [8192,2048] x [2048,2048]^T -> fp32 d_out
    dim3 g(DIM / 128, NTOK / 128);
    gemm_nt<float><<<g, 256, 0, stream>>>(attn16, wo16, (float*)d_out, NTOK, DIM, DIM);
  }
}
